// RobertaLayer_17188459119065
// MI455X (gfx1250) — matrix operand order repaired and checked by reading, not compiled, not run on hardware
//
#include <hip/hip_runtime.h>
#include <hip/hip_bf16.h>

#define S_LEN 4096
#define E_DIM 768
#define H_NUM 12
#define D_DIM 64
#define F_DIM 3072
#define WIN   128
#define GTOK  64
#define NCOL  448            // 64 global cols + 384 window cols
#define EPSV  1e-5f
#define SCALE 0.125f         // 1/sqrt(64)

typedef __attribute__((ext_vector_type(16))) __bf16 v16bf;
typedef __attribute__((ext_vector_type(8)))  float  v8f;

union Frag16 { v16bf v; uint4 q[2]; unsigned short h[16]; };

__device__ __forceinline__ unsigned short f2bf(float f) {
    unsigned u = __float_as_uint(f);
    u += 0x7FFFu + ((u >> 16) & 1u);          // round-to-nearest-even
    return (unsigned short)(u >> 16);
}

// ---- CDNA5 async global->LDS copy (ASYNCcnt-tracked, no VGPR data) ---------
__device__ __forceinline__ void async_load_b128(void* lds_dst, const void* gsrc) {
    unsigned ldsoff = (unsigned)(unsigned long long)lds_dst;   // LDS addr = low 32b
    asm volatile("global_load_async_to_lds_b128 %0, %1, off"
                 :: "v"(ldsoff), "v"((unsigned long long)gsrc) : "memory");
}
__device__ __forceinline__ void wait_async_le3() {
    asm volatile("s_wait_asynccnt 0x3" ::: "memory");
}
__device__ __forceinline__ void wait_async_0() {
    asm volatile("s_wait_asynccnt 0x0" ::: "memory");
}

// ---------------------------------------------------------------- fp32 -> bf16
__global__ void cvt_f32_to_bf16_kernel(const float* __restrict__ in,
                                       unsigned short* __restrict__ out, int n) {
    int i = blockIdx.x * blockDim.x + threadIdx.x;
    int stride = gridDim.x * blockDim.x;
    for (; i < n; i += stride) out[i] = f2bf(in[i]);
}

// ------------------------------------------- fp32 [R][C] -> bf16 [C][R] tiled
__global__ __launch_bounds__(256) void transpose_cvt_kernel(
    const float* __restrict__ in, unsigned short* __restrict__ out, int R, int Cc)
{
    __shared__ unsigned short t[32][33];
    const int c0 = blockIdx.x * 32, r0 = blockIdx.y * 32;
    const int tx = threadIdx.x & 31, ty = threadIdx.x >> 5;   // 32 x 8
    #pragma unroll
    for (int i = 0; i < 32; i += 8) {
        int r = r0 + ty + i, c = c0 + tx;
        t[ty + i][tx] = (r < R && c < Cc) ? f2bf(in[(long)r * Cc + c]) : 0;
    }
    __syncthreads();
    #pragma unroll
    for (int i = 0; i < 32; i += 8) {
        int c = c0 + ty + i, r = r0 + tx;
        if (c < Cc && r < R) out[(long)c * R + r] = t[tx][ty + i];
    }
}

// ------------------------------------------------- generic batched WMMA GEMM
// C[b] = act(A[b][MxK] * Bt[b][NxK]^T + bias). A row-major, Bt row-major [N][K].
// Optional bf16 output copy (Cbf), optionally transposed to [N][M] (bfT).
// Block tile 128(M) x 64(N), K-step 32, double-buffered async-to-LDS staging.
__global__ __launch_bounds__(256) void wmma_gemm_kernel(
    const unsigned short* __restrict__ A, long lda, long sAb,
    const unsigned short* __restrict__ Bt, long ldbt, long sBtb,
    const float* __restrict__ bias,
    float* __restrict__ C, unsigned short* __restrict__ Cbf,
    long ldc, long sCb,
    int M, int N, int K, int act, int bfT)
{
    __shared__ __align__(16) unsigned short sA[2][128 * 32];
    __shared__ __align__(16) unsigned short sB[2][64 * 32];
    const int tid  = threadIdx.x;
    const int wave = tid >> 5, lane = tid & 31;
    const int half = lane >> 4, l16 = lane & 15;
    const int n0 = blockIdx.x * 64;
    const int m0 = blockIdx.y * 128;
    A  += (long)blockIdx.z * sAb;
    Bt += (long)blockIdx.z * sBtb;
    if (C)   C   += (long)blockIdx.z * sCb;
    if (Cbf) Cbf += (long)blockIdx.z * sCb;

    const int r0 = tid >> 2, ko = (tid & 3) * 8;   // staging chunk per thread

    v8f acc[4];
    const v8f vzero = {0.f,0.f,0.f,0.f,0.f,0.f,0.f,0.f};
    #pragma unroll
    for (int j = 0; j < 4; ++j) acc[j] = vzero;

    // prime first tile (3 async b128 per thread)
    async_load_b128(&sA[0][r0 * 32 + ko],        A + (long)(m0 + r0) * lda + ko);
    async_load_b128(&sA[0][(r0 + 64) * 32 + ko], A + (long)(m0 + r0 + 64) * lda + ko);
    async_load_b128(&sB[0][r0 * 32 + ko],        Bt + (long)(n0 + r0) * ldbt + ko);

    int ib = 0;
    for (int kt = 0; kt < K; kt += 32, ib ^= 1) {
        if (kt + 32 < K) {
            // issue next tile into the other buffer, then wait only on current
            async_load_b128(&sA[ib ^ 1][r0 * 32 + ko],
                            A + (long)(m0 + r0) * lda + kt + 32 + ko);
            async_load_b128(&sA[ib ^ 1][(r0 + 64) * 32 + ko],
                            A + (long)(m0 + r0 + 64) * lda + kt + 32 + ko);
            async_load_b128(&sB[ib ^ 1][r0 * 32 + ko],
                            Bt + (long)(n0 + r0) * ldbt + kt + 32 + ko);
            wait_async_le3();
        } else {
            wait_async_0();
        }
        __syncthreads();

        Frag16 a;
        {
            const unsigned short* p = &sA[ib][(wave * 16 + l16) * 32 + half * 8];
            a.q[0] = *(const uint4*)(p);
            a.q[1] = *(const uint4*)(p + 16);
        }
        #pragma unroll
        for (int j = 0; j < 4; ++j) {
            Frag16 b;
            const unsigned short* p = &sB[ib][(j * 16 + l16) * 32 + half * 8];
            b.q[0] = *(const uint4*)(p);
            b.q[1] = *(const uint4*)(p + 16);
            acc[j] = __builtin_amdgcn_wmma_f32_16x16x32_bf16(
                false, a.v, false, b.v, (short)0, acc[j], false, false);
        }
        __syncthreads();
    }

    #pragma unroll
    for (int j = 0; j < 4; ++j) {
        int col = n0 + j * 16 + l16;
        float bs = bias ? bias[col] : 0.f;
        #pragma unroll
        for (int r = 0; r < 8; ++r) {
            int row = m0 + wave * 16 + r + half * 8;
            if (row < M) {
                float v = acc[j][r] + bs;
                if (act == 1) v = 0.5f * v * (1.f + erff(v * 0.70710678118f));
                if (C)   C[(long)row * ldc + col] = v;
                if (Cbf) {
                    if (bfT) Cbf[(long)col * M + row] = f2bf(v);
                    else     Cbf[(long)row * ldc + col] = f2bf(v);
                }
            }
        }
    }
}

// ------------------------------------------------------- banded attention
// One block per (chunk c, head h). Columns: [0,64)=global keys, [64,448)=window.
__global__ __launch_bounds__(256) void band_attn_kernel(
    const unsigned short* __restrict__ qb,
    const unsigned short* __restrict__ kb,
    const unsigned short* __restrict__ vb,
    const int* __restrict__ mask,
    float* __restrict__ ctx, unsigned short* __restrict__ ctxb)
{
    __shared__ __align__(16) float          sS [16 * NCOL];
    __shared__ __align__(16) unsigned short sP [16 * NCOL];
    __shared__ __align__(16) unsigned short sVt[64 * NCOL];   // [d][col]
    const int c = blockIdx.x, h = blockIdx.y;
    const int tid  = threadIdx.x;
    const int wave = tid >> 5, lane = tid & 31;
    const int half = lane >> 4, l16 = lane & 15;
    const int j0 = c * WIN - WIN;
    const v8f vzero = {0.f,0.f,0.f,0.f,0.f,0.f,0.f,0.f};

    // stage transposed V window once per block
    for (int col = tid; col < NCOL; col += 256) {
        int j = (col < GTOK) ? col : (j0 + col - GTOK);
        if (j >= 0 && j < S_LEN) {
            const unsigned short* src = vb + (long)j * E_DIM + h * D_DIM;
            for (int d = 0; d < 64; ++d) sVt[d * NCOL + col] = src[d];
        } else {
            for (int d = 0; d < 64; ++d) sVt[d * NCOL + col] = 0;
        }
    }
    __syncthreads();

    for (int strip = 0; strip < 8; ++strip) {
        const int srow0 = c * WIN + strip * 16;
        // ---- phase 1: scores = scale * Q Kt  (28 col-tiles, K=64)
        for (int t = wave; t < 28; t += 8) {
            v8f acc = vzero;
            #pragma unroll
            for (int kk = 0; kk < 64; kk += 32) {
                Frag16 a, b;
                const unsigned short* pa =
                    qb + (long)(srow0 + l16) * E_DIM + h * D_DIM + kk + half * 8;
                a.q[0] = *(const uint4*)(pa);
                a.q[1] = *(const uint4*)(pa + 16);
                int col = t * 16 + l16;
                int j = (col < GTOK) ? col : (j0 + col - GTOK);
                if (j >= 0 && j < S_LEN) {
                    const unsigned short* pb =
                        kb + (long)j * E_DIM + h * D_DIM + kk + half * 8;
                    b.q[0] = *(const uint4*)(pb);
                    b.q[1] = *(const uint4*)(pb + 16);
                } else {
                    uint4 z = {0u,0u,0u,0u}; b.q[0] = z; b.q[1] = z;
                }
                acc = __builtin_amdgcn_wmma_f32_16x16x32_bf16(
                    false, a.v, false, b.v, (short)0, acc, false, false);
            }
            #pragma unroll
            for (int r = 0; r < 8; ++r)
                sS[(r + half * 8) * NCOL + t * 16 + l16] = acc[r] * SCALE;
        }
        __syncthreads();
        // ---- phase 2: masked softmax (16 threads per row)
        {
            const int r = tid >> 4, g = tid & 15;
            const int s = srow0 + r;
            float m = -3.0e38f;
            for (int col = g; col < NCOL; col += 16) {
                bool ok;
                if (col < GTOK) {
                    ok = (mask[col] > 0);
                } else {
                    int j = j0 + col - GTOK;
                    ok = (j >= GTOK) && (j < S_LEN) &&
                         (j >= s - WIN) && (j <= s + WIN) && (mask[j] > 0);
                }
                float v = ok ? sS[r * NCOL + col] : -1.0e9f;
                sS[r * NCOL + col] = v;
                m = fmaxf(m, v);
            }
            for (int o = 8; o > 0; o >>= 1) m = fmaxf(m, __shfl_xor(m, o, 32));
            float l = 0.f;
            for (int col = g; col < NCOL; col += 16) {
                float p = __expf(sS[r * NCOL + col] - m);
                sS[r * NCOL + col] = p;
                l += p;
            }
            for (int o = 8; o > 0; o >>= 1) l += __shfl_xor(l, o, 32);
            float inv = 1.f / l;
            for (int col = g; col < NCOL; col += 16)
                sP[r * NCOL + col] = f2bf(sS[r * NCOL + col] * inv);
        }
        __syncthreads();
        // ---- phase 3: out = P * Vwin  (waves 0..3 -> d-tiles, 14 K-steps)
        if (wave < 4) {
            v8f acc = vzero;
            for (int ks = 0; ks < NCOL; ks += 32) {
                Frag16 a, b;
                const unsigned short* pa = &sP[l16 * NCOL + ks + half * 8];
                a.q[0] = *(const uint4*)(pa);
                a.q[1] = *(const uint4*)(pa + 16);
                const unsigned short* pb =
                    &sVt[(wave * 16 + l16) * NCOL + ks + half * 8];
                b.q[0] = *(const uint4*)(pb);
                b.q[1] = *(const uint4*)(pb + 16);
                acc = __builtin_amdgcn_wmma_f32_16x16x32_bf16(
                    false, a.v, false, b.v, (short)0, acc, false, false);
            }
            #pragma unroll
            for (int r = 0; r < 8; ++r) {
                int row = srow0 + r + half * 8;
                int d   = wave * 16 + l16;
                float v = acc[r];
                ctx [(long)row * E_DIM + h * D_DIM + d] = v;
                ctxb[(long)row * E_DIM + h * D_DIM + d] = f2bf(v);
            }
        }
        __syncthreads();
    }
}

// ------------------------------------- global-token attention probabilities
__global__ __launch_bounds__(256) void global_probs_kernel(
    const float* __restrict__ qg, const float* __restrict__ kg,
    const int* __restrict__ mask, unsigned short* __restrict__ Pb)
{
    __shared__ float sQ[64];
    __shared__ float red[256];
    const int rowid = blockIdx.x;                 // h*64 + g
    const int h = rowid >> 6, g = rowid & 63;
    const int tid = threadIdx.x;
    if (tid < 64) sQ[tid] = qg[(long)g * E_DIM + h * D_DIM + tid];
    __syncthreads();

    float sc[16];
    float m = -3.0e38f;
    #pragma unroll
    for (int i = 0; i < 16; ++i) {
        int s = tid + i * 256;
        const float* kr = kg + (long)s * E_DIM + h * D_DIM;
        float dot = 0.f;
        for (int d = 0; d < 64; ++d) dot += sQ[d] * kr[d];
        float v = (mask[s] > 0) ? dot * SCALE : -1.0e9f;
        sc[i] = v;
        m = fmaxf(m, v);
    }
    red[tid] = m; __syncthreads();
    for (int o = 128; o > 0; o >>= 1) {
        if (tid < o) red[tid] = fmaxf(red[tid], red[tid + o]);
        __syncthreads();
    }
    m = red[0]; __syncthreads();
    float l = 0.f;
    #pragma unroll
    for (int i = 0; i < 16; ++i) { sc[i] = __expf(sc[i] - m); l += sc[i]; }
    red[tid] = l; __syncthreads();
    for (int o = 128; o > 0; o >>= 1) {
        if (tid < o) red[tid] += red[tid + o];
        __syncthreads();
    }
    float inv = 1.f / red[0];
    #pragma unroll
    for (int i = 0; i < 16; ++i)
        Pb[(long)rowid * S_LEN + tid + i * 256] = f2bf(sc[i] * inv);
}

// ------------------------------------------------------ LayerNorm + residual
__global__ __launch_bounds__(256) void ln_residual_kernel(
    const float* __restrict__ a, const float* __restrict__ res,
    const float* __restrict__ gamma, const float* __restrict__ beta,
    float* __restrict__ out, unsigned short* __restrict__ outb, int cols)
{
    __shared__ float r1[256], r2[256];
    const int row = blockIdx.x, tid = threadIdx.x;
    const float* pa = a   + (long)row * cols;
    const float* pr = res + (long)row * cols;
    float s = 0.f, s2 = 0.f;
    for (int i = tid; i < cols; i += 256) {
        float v = pa[i] + pr[i]; s += v; s2 += v * v;
    }
    r1[tid] = s; r2[tid] = s2; __syncthreads();
    for (int o = 128; o > 0; o >>= 1) {
        if (tid < o) { r1[tid] += r1[tid + o]; r2[tid] += r2[tid + o]; }
        __syncthreads();
    }
    float mean = r1[0] / (float)cols;
    float var  = r2[0] / (float)cols - mean * mean;
    float inv  = rsqrtf(var + EPSV);
    for (int i = tid; i < cols; i += 256) {
        float v = (pa[i] + pr[i] - mean) * inv * gamma[i] + beta[i];
        out[(long)row * cols + i] = v;
        if (outb) outb[(long)row * cols + i] = f2bf(v);
    }
}

// ============================================================================
extern "C" void kernel_launch(void* const* d_in, const int* in_sizes, int n_in,
                              void* d_out, int out_size, void* d_ws, size_t ws_size,
                              hipStream_t stream)
{
    (void)in_sizes; (void)n_in; (void)out_size; (void)ws_size;
    const float* x    = (const float*)d_in[0];
    const int*   mask = (const int*)  d_in[1];
    const float* Wq  = (const float*)d_in[2];  const float* bq  = (const float*)d_in[3];
    const float* Wk  = (const float*)d_in[4];  const float* bk  = (const float*)d_in[5];
    const float* Wv  = (const float*)d_in[6];  const float* bv  = (const float*)d_in[7];
    const float* Wqg = (const float*)d_in[8];  const float* bqg = (const float*)d_in[9];
    const float* Wkg = (const float*)d_in[10]; const float* bkg = (const float*)d_in[11];
    const float* Wvg = (const float*)d_in[12]; const float* bvg = (const float*)d_in[13];
    const float* Wo  = (const float*)d_in[14]; const float* bo  = (const float*)d_in[15];
    const float* g1  = (const float*)d_in[16]; const float* b1  = (const float*)d_in[17];
    const float* Wi  = (const float*)d_in[18]; const float* bi  = (const float*)d_in[19];
    const float* Wo2 = (const float*)d_in[20]; const float* bo2 = (const float*)d_in[21];
    const float* g2  = (const float*)d_in[22]; const float* b2  = (const float*)d_in[23];

    char* ws = (char*)d_ws;
    size_t off = 0;
    auto alloc = [&](size_t bytes) -> void* {
        void* p = ws + off;
        off = (off + bytes + 255) & ~(size_t)255;
        return p;
    };
    const size_t SE = (size_t)S_LEN * E_DIM;
    const size_t SF = (size_t)S_LEN * F_DIM;
    const size_t EE = (size_t)E_DIM * E_DIM;

    unsigned short* xb    = (unsigned short*)alloc(SE * 2);
    unsigned short* wqT   = (unsigned short*)alloc(EE * 2);
    unsigned short* wkT   = (unsigned short*)alloc(EE * 2);
    unsigned short* wvT   = (unsigned short*)alloc(EE * 2);
    unsigned short* wqgT  = (unsigned short*)alloc(EE * 2);
    unsigned short* wkgT  = (unsigned short*)alloc(EE * 2);
    unsigned short* wvgT  = (unsigned short*)alloc(EE * 2);
    unsigned short* woT   = (unsigned short*)alloc(EE * 2);
    unsigned short* wiT   = (unsigned short*)alloc((size_t)E_DIM * F_DIM * 2);
    unsigned short* wo2T  = (unsigned short*)alloc((size_t)F_DIM * E_DIM * 2);
    float* kgf  = (float*)alloc(SE * 4);
    float* qgf  = (float*)alloc((size_t)GTOK * E_DIM * 4);
    unsigned short* qbb  = (unsigned short*)alloc(SE * 2);
    unsigned short* kbb  = (unsigned short*)alloc(SE * 2);
    unsigned short* vbb  = (unsigned short*)alloc(SE * 2);
    unsigned short* vgT  = (unsigned short*)alloc(SE * 2);        // [E][S]
    unsigned short* Pb   = (unsigned short*)alloc((size_t)H_NUM * GTOK * S_LEN * 2);
    float* ctx      = (float*)alloc(SE * 4);
    unsigned short* ctxb = (unsigned short*)alloc(SE * 2);
    float* lin1     = (float*)alloc(SE * 4);
    float* attn_out = (float*)alloc(SE * 4);
    unsigned short* attn_outb = (unsigned short*)alloc(SE * 2);
    unsigned short* interb    = (unsigned short*)alloc(SF * 2);
    float* ffn2     = (float*)alloc(SE * 4);

    auto cvt = [&](const float* src, unsigned short* dst, size_t n) {
        int blocks = (int)((n + 1023) / 1024); if (blocks > 4096) blocks = 4096;
        cvt_f32_to_bf16_kernel<<<blocks, 256, 0, stream>>>(src, dst, (int)n);
    };
    auto tc = [&](const float* src, unsigned short* dst, int R, int Cc) {
        transpose_cvt_kernel<<<dim3((Cc + 31) / 32, (R + 31) / 32), 256, 0, stream>>>(
            src, dst, R, Cc);
    };
    auto gemm = [&](const unsigned short* A, long lda, long sAb,
                    const unsigned short* Bt, long ldbt, long sBtb,
                    const float* bias, float* C, unsigned short* Cbf,
                    long ldc, long sCb, int M, int N, int K, int act, int bfT,
                    int batch) {
        dim3 grid(N / 64, (M + 127) / 128, batch);
        wmma_gemm_kernel<<<grid, 256, 0, stream>>>(A, lda, sAb, Bt, ldbt, sBtb,
                                                   bias, C, Cbf, ldc, sCb,
                                                   M, N, K, act, bfT);
    };

    // 1) bf16 input + transposed bf16 weights [N][K]
    cvt(x, xb, SE);
    tc(Wq,  wqT,  E_DIM, E_DIM);  tc(Wk,  wkT,  E_DIM, E_DIM);
    tc(Wv,  wvT,  E_DIM, E_DIM);  tc(Wqg, wqgT, E_DIM, E_DIM);
    tc(Wkg, wkgT, E_DIM, E_DIM);  tc(Wvg, wvgT, E_DIM, E_DIM);
    tc(Wo,  woT,  E_DIM, E_DIM);
    tc(Wi,  wiT,  E_DIM, F_DIM);                 // -> [F][E]
    tc(Wo2, wo2T, F_DIM, E_DIM);                 // -> [E][F]

    // 2) projections (bf16 outputs written directly from the epilogue)
    gemm(xb, E_DIM, 0, wqT,  E_DIM, 0, bq,  nullptr, qbb, E_DIM, 0,
         S_LEN, E_DIM, E_DIM, 0, 0, 1);
    gemm(xb, E_DIM, 0, wkT,  E_DIM, 0, bk,  nullptr, kbb, E_DIM, 0,
         S_LEN, E_DIM, E_DIM, 0, 0, 1);
    gemm(xb, E_DIM, 0, wvT,  E_DIM, 0, bv,  nullptr, vbb, E_DIM, 0,
         S_LEN, E_DIM, E_DIM, 0, 0, 1);
    gemm(xb, E_DIM, 0, wkgT, E_DIM, 0, bkg, kgf, nullptr, E_DIM, 0,
         S_LEN, E_DIM, E_DIM, 0, 0, 1);
    gemm(xb, E_DIM, 0, wvgT, E_DIM, 0, bvg, nullptr, vgT, E_DIM, 0,
         S_LEN, E_DIM, E_DIM, 0, 1 /*bfT -> [E][S]*/, 1);
    gemm(xb, E_DIM, 0, wqgT, E_DIM, 0, bqg, qgf, nullptr, E_DIM, 0,
         GTOK,  E_DIM, E_DIM, 0, 0, 1);

    // 3) banded + global-key attention (writes all ctx rows, fp32 + bf16)
    band_attn_kernel<<<dim3(32, 12), 256, 0, stream>>>(qbb, kbb, vbb, mask, ctx, ctxb);

    // 4) global-token attention: probs, then batched P*V WMMA GEMM (rows 0..63)
    global_probs_kernel<<<dim3(H_NUM * GTOK), 256, 0, stream>>>(qgf, kgf, mask, Pb);
    gemm(Pb, S_LEN, (long)GTOK * S_LEN, vgT, S_LEN, (long)D_DIM * S_LEN,
         nullptr, ctx, ctxb, E_DIM, D_DIM, GTOK, D_DIM, S_LEN, 0, 0, H_NUM);

    // 5) output projection + LN(residual x)
    gemm(ctxb, E_DIM, 0, woT, E_DIM, 0, bo, lin1, nullptr, E_DIM, 0,
         S_LEN, E_DIM, E_DIM, 0, 0, 1);
    ln_residual_kernel<<<S_LEN, 256, 0, stream>>>(lin1, x, g1, b1,
                                                  attn_out, attn_outb, E_DIM);

    // 6) FFN (exact GELU fused), then LN(residual attn_out) -> d_out
    gemm(attn_outb, E_DIM, 0, wiT, E_DIM, 0, bi, nullptr, interb, F_DIM, 0,
         S_LEN, F_DIM, E_DIM, 1, 0, 1);
    gemm(interb, F_DIM, 0, wo2T, F_DIM, 0, bo2, ffn2, nullptr, E_DIM, 0,
         S_LEN, E_DIM, F_DIM, 0, 0, 1);
    ln_residual_kernel<<<S_LEN, 256, 0, stream>>>(ffn2, attn_out, g2, b2,
                                                  (float*)d_out, nullptr, E_DIM);
}
